// SC2FusionBlock_58153857188055
// MI455X (gfx1250) — matrix operand order repaired and checked by reading, not compiled, not run on hardware
//
#include <hip/hip_runtime.h>
#include <math.h>

// ---------------------------------------------------------------------------
// Types for CDNA5 WMMA (wave32): 16x16x32 bf16 -> f32
// ---------------------------------------------------------------------------
typedef __attribute__((ext_vector_type(16))) __bf16 v16bf;
typedef __attribute__((ext_vector_type(16))) float  v16f;
typedef __attribute__((ext_vector_type(8)))  float  v8f;

union F16V { v16f v; float4 q[4]; };

__device__ __forceinline__ v16bf cvt16(v16f f) {
  return __builtin_convertvector(f, v16bf);   // native packed f32->bf16 cvts
}

#define WMMA_BF16(A, B, C) \
  __builtin_amdgcn_wmma_f32_16x16x32_bf16(false, (A), false, (B), (short)0, (C), false, false)

// Async global->LDS copy (CDNA5, ASYNCcnt-tracked). laddr = LDS byte offset,
// gp = per-lane global address.
__device__ __forceinline__ void async_ld_b32(unsigned laddr, const float* gp) {
  asm volatile("global_load_async_to_lds_b32 %0, %1, off" :: "v"(laddr), "v"(gp) : "memory");
}
__device__ __forceinline__ void async_ld_b128(unsigned laddr, const float* gp) {
  asm volatile("global_load_async_to_lds_b128 %0, %1, off" :: "v"(laddr), "v"(gp) : "memory");
}
__device__ __forceinline__ void async_wait0() {
  asm volatile("s_wait_asynccnt 0x0" ::: "memory");
}
__device__ __forceinline__ unsigned lds_off(const void* p) {
  return (unsigned)(uintptr_t)p;              // LDS aperture: addr[31:0] = LDS offset
}

// ---------------------------------------------------------------------------
// Pooling: tp[b,c] = mean_n feat[b,n,c]; cond[b] = mean_n ||feat[b,n,:]||
// ---------------------------------------------------------------------------
__global__ void __launch_bounds__(256)
pool_kernel(const float* __restrict__ feat, float* __restrict__ tp,
            float* __restrict__ cond, int N, int C) {
  const int b = blockIdx.x;
  const int c = threadIdx.x;               // C == 256
  float s = 0.f;
  for (int n = 0; n < N; ++n)
    s += feat[((size_t)b * N + n) * C + c];
  tp[b * C + c] = s / (float)N;

  float acc = 0.f;
  for (int n = c; n < N; n += 256) {
    const float* row = feat + ((size_t)b * N + n) * C;
    float ss = 0.f;
    for (int k = 0; k < C; ++k) { float v = row[k]; ss += v * v; }
    acc += sqrtf(ss);
  }
  __shared__ float red[256];
  red[c] = acc;
  __syncthreads();
  for (int off = 128; off > 0; off >>= 1) {
    if (c < off) red[c] += red[c + off];
    __syncthreads();
  }
  if (c == 0) cond[b] = red[0] / (float)N;
}

// ---------------------------------------------------------------------------
// Small MLP: bias[b,h] = relu([tp,sp,tm] @ Wc1 + bc1) @ Wc2 + bc2
//            gate[b,c] = sigmoid(tp @ Wg + bg)
// ---------------------------------------------------------------------------
__global__ void __launch_bounds__(256)
mlp_kernel(const float* __restrict__ tp, const float* __restrict__ sp,
           const float* __restrict__ tm,
           const float* __restrict__ Wc1, const float* __restrict__ bc1,
           const float* __restrict__ Wc2, const float* __restrict__ bc2,
           const float* __restrict__ Wg,  const float* __restrict__ bg,
           float* __restrict__ biasOut, float* __restrict__ gateOut) {
  const int C = 256, H = 8;
  const int b = blockIdx.x, j = threadIdx.x;
  __shared__ float tps[256];
  __shared__ float hs[256];
  tps[j] = tp[b * C + j];
  __syncthreads();

  float h = bc1[j];
  for (int k = 0; k < C; ++k) h += tps[k] * Wc1[k * C + j];
  h += sp[b] * Wc1[C * C + j];
  h += tm[b] * Wc1[(C + 1) * C + j];
  hs[j] = fmaxf(h, 0.f);

  float g = bg[j];
  for (int k = 0; k < C; ++k) g += tps[k] * Wg[k * C + j];
  gateOut[b * C + j] = 1.f / (1.f + __expf(-g));
  __syncthreads();

  if (j < H) {
    float o = bc2[j];
    for (int k = 0; k < C; ++k) o += hs[k] * Wc2[k * H + j];
    biasOut[b * H + j] = o;
  }
}

// ---------------------------------------------------------------------------
// WMMA GEMM: Y[M,256] = X[M,256] @ W[256,256] + bias
// 256 threads = 8 waves; wave w -> rows [blk*128 + 16w, +16), cols [nb, nb+64)
// W tile (32x64) async-copied into LDS as f32 [n][k] (scatter-transpose) so
// B-fragments are contiguous ds_load_b128 runs.
// ---------------------------------------------------------------------------
__global__ void __launch_bounds__(256)
gemm_bias(const float* __restrict__ X, const float* __restrict__ W,
          const float* __restrict__ bias, float* __restrict__ Y) {
  __shared__ __align__(16) float wtile[64 * 32];   // [n][k] f32

  const int tid  = threadIdx.x;
  const int wave = tid >> 5;
  const int lane = tid & 31;
  const int rowbase = blockIdx.x * 128 + wave * 16;
  const int nb      = blockIdx.y * 64;

  v8f acc[4];
  for (int t = 0; t < 4; ++t)
    for (int i = 0; i < 8; ++i) acc[t][i] = 0.f;

  const int arow = rowbase + (lane & 15);
  const int kb   = (lane >> 4) * 16;      // A-layout K offset per half-wave
  const int nlo  = lane & 15;
  const int kst  = (lane >> 4) * 16;      // B-layout K offset per half-wave
  const unsigned lbase = lds_off(&wtile[0]);
  const int wn  = tid & 63;               // cooperative copy: column
  const int wk0 = tid >> 6;               // and starting k-row (0..3)

  for (int kk = 0; kk < 256; kk += 32) {
    // async scatter-transpose W[kk+kl][nb+n] -> wtile[n*32 + kl]
    for (int kl = wk0; kl < 32; kl += 4)
      async_ld_b32(lbase + (unsigned)((wn * 32 + kl) * 4),
                   W + (size_t)(kk + kl) * 256 + nb + wn);
    async_wait0();
    __syncthreads();

    // A fragment (contiguous k = 16h + s, matching B)
    const float* xp = X + (size_t)arow * 256 + kk;
    F16V af;
    af.q[0] = *(const float4*)(xp + kb);
    af.q[1] = *(const float4*)(xp + kb + 4);
    af.q[2] = *(const float4*)(xp + kb + 8);
    af.q[3] = *(const float4*)(xp + kb + 12);
    v16bf a = cvt16(af.v);

    if (kk + 32 < 256) __builtin_prefetch(xp + 32, 0, 0);  // global_prefetch_b8

    for (int t = 0; t < 4; ++t) {
      const float* wp = &wtile[(t * 16 + nlo) * 32 + kst];
      F16V bfv;
      bfv.q[0] = *(const float4*)(wp);
      bfv.q[1] = *(const float4*)(wp + 4);
      bfv.q[2] = *(const float4*)(wp + 8);
      bfv.q[3] = *(const float4*)(wp + 12);
      acc[t] = WMMA_BF16(a, cvt16(bfv.v), acc[t]);
    }
    __syncthreads();
  }

  // epilogue: C layout -> VGPR i: row = rowbase + i + 8*(lane>=16)
  const int rhalf = (lane >> 4) * 8;
  for (int t = 0; t < 4; ++t) {
    const int col = nb + t * 16 + nlo;
    const float bv = bias[col];
    for (int i = 0; i < 8; ++i)
      Y[(size_t)(rowbase + i + rhalf) * 256 + col] = acc[t][i] + bv;
  }
}

// ---------------------------------------------------------------------------
// Flash attention per (b,h): O = softmax(Q K^T * scale + bias) V
// Q:[B,Nq,256] K,V:[B,Nk,256] (head h in cols h*32..h*32+31). 1 wave / 16 rows.
// K chunk (32x32) async-staged [key][d]; V chunk async-staged transposed
// [d][key] so both B-fragments read contiguous LDS runs.
// ---------------------------------------------------------------------------
__global__ void __launch_bounds__(128)
flash_attn(const float* __restrict__ Q, const float* __restrict__ Kf,
           const float* __restrict__ Vf, const float* __restrict__ bias,
           float* __restrict__ O, int Nq, int Nk, float scale) {
  const int C = 256, H = 8, D = 32;
  __shared__ __align__(16) float klds[32 * 32];     // [key][d]
  __shared__ __align__(16) float vlds[32 * 32];     // [d][key]
  __shared__ __align__(16) float psh[4][16 * 32];

  const int wave = threadIdx.x >> 5, lane = threadIdx.x & 31;
  const int h = blockIdx.y, b = blockIdx.z;
  const int qbase = (blockIdx.x * 4 + wave) * 16;
  float* pl = psh[wave];

  const float bsc = bias[b * H + h];
  const int nlo = lane & 15;
  const int kb  = (lane >> 4) * 16;
  const int khalf = (lane >> 4) * 16;
  const int rhalf = (lane >> 4) * 8;
  const unsigned kbase = lds_off(&klds[0]);
  const unsigned vbase = lds_off(&vlds[0]);

  // Q fragment (loaded once: d == 32 == one WMMA K step)
  const float* qp = Q + ((size_t)b * Nq + qbase + nlo) * C + h * D;
  F16V qf;
  qf.q[0] = *(const float4*)(qp + kb);
  qf.q[1] = *(const float4*)(qp + kb + 4);
  qf.q[2] = *(const float4*)(qp + kb + 8);
  qf.q[3] = *(const float4*)(qp + kb + 12);
  const v16bf qfrag = cvt16(qf.v);

  v8f o0, o1;
  float m[8], l[8];
  for (int i = 0; i < 8; ++i) { o0[i] = 0.f; o1[i] = 0.f; m[i] = -1e30f; l[i] = 0.f; }
  const v8f vzero = o0;

  for (int kc = 0; kc < Nk; kc += 32) {
    // --- cooperative async staging of K (b128) and V (scatter-transpose b32)
    const size_t rowb = ((size_t)b * Nk + kc) * C + h * D;
    for (int c4 = threadIdx.x; c4 < 256; c4 += 128) {       // 32 keys x 8 float4
      const int key = c4 >> 3, d4 = (c4 & 7) * 4;
      async_ld_b128(kbase + (unsigned)((key * 32 + d4) * 4),
                    Kf + rowb + (size_t)key * C + d4);
    }
    for (int idx = threadIdx.x; idx < 1024; idx += 128) {   // V transposed
      const int key = idx >> 5, d = idx & 31;
      async_ld_b32(vbase + (unsigned)((d * 32 + key) * 4),
                   Vf + rowb + (size_t)key * C + d);
    }
    async_wait0();
    __syncthreads();

    // --- S = Q K^T for 32 keys (two 16-key B fragments from LDS) ---
    F16V k0, k1;
    {
      const float* kp0 = &klds[nlo * 32 + khalf];
      const float* kp1 = &klds[(16 + nlo) * 32 + khalf];
      k0.q[0] = *(const float4*)(kp0);      k0.q[1] = *(const float4*)(kp0 + 4);
      k0.q[2] = *(const float4*)(kp0 + 8);  k0.q[3] = *(const float4*)(kp0 + 12);
      k1.q[0] = *(const float4*)(kp1);      k1.q[1] = *(const float4*)(kp1 + 4);
      k1.q[2] = *(const float4*)(kp1 + 8);  k1.q[3] = *(const float4*)(kp1 + 12);
    }
    v8f s0 = WMMA_BF16(qfrag, cvt16(k0.v), vzero);
    v8f s1 = WMMA_BF16(qfrag, cvt16(k1.v), vzero);

    // --- online softmax over this 32-key chunk ---
    for (int i = 0; i < 8; ++i) {
      float a0 = s0[i] * scale + bsc;
      float a1 = s1[i] * scale + bsc;
      float cm = fmaxf(a0, a1);
      cm = fmaxf(cm, __shfl_xor(cm, 8, 32));
      cm = fmaxf(cm, __shfl_xor(cm, 4, 32));
      cm = fmaxf(cm, __shfl_xor(cm, 2, 32));
      cm = fmaxf(cm, __shfl_xor(cm, 1, 32));
      float mn = fmaxf(m[i], cm);
      float f  = __expf(m[i] - mn);
      float p0 = __expf(a0 - mn);
      float p1 = __expf(a1 - mn);
      float rs = p0 + p1;
      rs += __shfl_xor(rs, 8, 32);
      rs += __shfl_xor(rs, 4, 32);
      rs += __shfl_xor(rs, 2, 32);
      rs += __shfl_xor(rs, 1, 32);
      l[i] = l[i] * f + rs;
      m[i] = mn;
      o0[i] *= f;
      o1[i] *= f;
      const int row = i + rhalf;                 // C-layout row for this half
      pl[row * 32 + nlo]      = p0;
      pl[row * 32 + 16 + nlo] = p1;
    }
    __syncthreads();

    // --- P fragment (A layout) + V fragments (B layout) from LDS ---
    F16V pf, v0, v1;
    {
      const float* pr = &pl[nlo * 32];
      pf.q[0] = *(const float4*)(pr + kb);
      pf.q[1] = *(const float4*)(pr + kb + 4);
      pf.q[2] = *(const float4*)(pr + kb + 8);
      pf.q[3] = *(const float4*)(pr + kb + 12);
      const float* vp0 = &vlds[nlo * 32 + khalf];
      const float* vp1 = &vlds[(16 + nlo) * 32 + khalf];
      v0.q[0] = *(const float4*)(vp0);      v0.q[1] = *(const float4*)(vp0 + 4);
      v0.q[2] = *(const float4*)(vp0 + 8);  v0.q[3] = *(const float4*)(vp0 + 12);
      v1.q[0] = *(const float4*)(vp1);      v1.q[1] = *(const float4*)(vp1 + 4);
      v1.q[2] = *(const float4*)(vp1 + 8);  v1.q[3] = *(const float4*)(vp1 + 12);
    }
    v16bf pfrag = cvt16(pf.v);
    o0 = WMMA_BF16(pfrag, cvt16(v0.v), o0);
    o1 = WMMA_BF16(pfrag, cvt16(v1.v), o1);
    __syncthreads();
  }

  // --- epilogue: normalize and write [B,Nq,256] with head cols h*32.. ---
  for (int i = 0; i < 8; ++i) {
    const float inv = 1.f / l[i];
    float* op = O + ((size_t)b * Nq + qbase + i + rhalf) * C + h * D;
    op[nlo]      = o0[i] * inv;
    op[16 + nlo] = o1[i] * inv;
  }
}

// ---------------------------------------------------------------------------
// out = g * proj + (1-g) * orig, g broadcast over tokens
// ---------------------------------------------------------------------------
__global__ void __launch_bounds__(256)
gate_blend(const float* __restrict__ proj, const float* __restrict__ gate,
           const float* __restrict__ orig, float* __restrict__ out,
           int total, int NC, int C) {
  int idx = blockIdx.x * blockDim.x + threadIdx.x;
  if (idx >= total) return;
  int b = idx / NC;
  int c = idx % C;
  float g = gate[b * C + c];
  out[idx] = g * proj[idx] + (1.f - g) * orig[idx];
}

// ---------------------------------------------------------------------------
// Host-side orchestration
// ---------------------------------------------------------------------------
extern "C" void kernel_launch(void* const* d_in, const int* in_sizes, int n_in,
                              void* d_out, int out_size, void* d_ws, size_t ws_size,
                              hipStream_t stream) {
  const int B = 16, Nv = 1024, Na = 512, C = 256, H = 8;
  const float scale = 0.17677669529663687f; // 32^-0.5

  const float* image = (const float*)d_in[0];
  const float* audio = (const float*)d_in[1];
  const float *Wq_i=(const float*)d_in[2],  *bq_i=(const float*)d_in[3];
  const float *Wk_i=(const float*)d_in[4],  *bk_i=(const float*)d_in[5];
  const float *Wv_i=(const float*)d_in[6],  *bv_i=(const float*)d_in[7];
  const float *Wo_i=(const float*)d_in[8],  *bo_i=(const float*)d_in[9];
  const float *Wq_a=(const float*)d_in[10], *bq_a=(const float*)d_in[11];
  const float *Wk_a=(const float*)d_in[12], *bk_a=(const float*)d_in[13];
  const float *Wv_a=(const float*)d_in[14], *bv_a=(const float*)d_in[15];
  const float *Wo_a=(const float*)d_in[16], *bo_a=(const float*)d_in[17];
  const float *Wc1_i=(const float*)d_in[18],*bc1_i=(const float*)d_in[19];
  const float *Wc2_i=(const float*)d_in[20],*bc2_i=(const float*)d_in[21];
  const float *Wc1_a=(const float*)d_in[22],*bc1_a=(const float*)d_in[23];
  const float *Wc2_a=(const float*)d_in[24],*bc2_a=(const float*)d_in[25];
  const float *Wg_i=(const float*)d_in[26], *bg_i=(const float*)d_in[27];
  const float *Wg_a=(const float*)d_in[28], *bg_a=(const float*)d_in[29];

  float* out_img = (float*)d_out;                         // [16,1024,256]
  float* out_aud = (float*)d_out + (size_t)B * Nv * C;    // [16,512,256]

  const size_t IMGSZ = (size_t)B * Nv * C;  // 4,194,304
  const size_t AUDSZ = (size_t)B * Na * C;  // 2,097,152
  float* ws = (float*)d_ws;
  float* bufA   = ws;                       // 16MB : Ki -> Qi -> tmp_i
  float* bufB   = bufA + IMGSZ;             // 16MB : Vi -> Va
  float* bufE   = bufB + IMGSZ;             // 16MB : tmp_a -> fused_i
  float* bufC   = bufE + IMGSZ;             //  8MB : Qa -> Ka
  float* bufD   = bufC + AUDSZ;             //  8MB : fused_a
  float* tp_img = bufD + AUDSZ;
  float* tp_aud = tp_img + B * C;
  float* sp     = tp_aud + B * C;           // spatial_cond [B]
  float* tm     = sp + B;                   // time_cond    [B]
  float* bias_a = tm + B;                   // [B,H]
  float* bias_i = bias_a + B * H;
  float* gate_a = bias_i + B * H;           // [B,C]
  float* gate_i = gate_a + B * C;

  // 1) pooled stats
  pool_kernel<<<dim3(B), dim3(256), 0, stream>>>(image, tp_img, sp, Nv, C);
  pool_kernel<<<dim3(B), dim3(256), 0, stream>>>(audio, tp_aud, tm, Na, C);

  // 2) conditioning MLPs + gates
  mlp_kernel<<<dim3(B), dim3(256), 0, stream>>>(tp_aud, sp, tm, Wc1_a, bc1_a,
      Wc2_a, bc2_a, Wg_a, bg_a, bias_a, gate_a);
  mlp_kernel<<<dim3(B), dim3(256), 0, stream>>>(tp_img, sp, tm, Wc1_i, bc1_i,
      Wc2_i, bc2_i, Wg_i, bg_i, bias_i, gate_i);

  // 3) phase 1 projections: Qa, Ki, Vi
  gemm_bias<<<dim3((B*Na)/128, 4), dim3(256), 0, stream>>>(audio, Wq_a, bq_a, bufC);
  gemm_bias<<<dim3((B*Nv)/128, 4), dim3(256), 0, stream>>>(image, Wk_i, bk_i, bufA);
  gemm_bias<<<dim3((B*Nv)/128, 4), dim3(256), 0, stream>>>(image, Wv_i, bv_i, bufB);

  // 4) audio attends over image -> fused_a (pre out-proj)
  flash_attn<<<dim3(Na/64, H, B), dim3(128), 0, stream>>>(bufC, bufA, bufB,
      bias_a, bufD, Na, Nv, scale);

  // 5) out-proj + gate -> audio_out
  gemm_bias<<<dim3((B*Na)/128, 4), dim3(256), 0, stream>>>(bufD, Wo_a, bo_a, bufE);
  gate_blend<<<dim3((unsigned)((AUDSZ + 255) / 256)), dim3(256), 0, stream>>>(
      bufE, gate_a, audio, out_aud, (int)AUDSZ, Na * C, C);

  // 6) phase 2 projections: Qi, Ka, Va (K/V from UPDATED audio)
  gemm_bias<<<dim3((B*Nv)/128, 4), dim3(256), 0, stream>>>(image,   Wq_i, bq_i, bufA);
  gemm_bias<<<dim3((B*Na)/128, 4), dim3(256), 0, stream>>>(out_aud, Wk_a, bk_a, bufC);
  gemm_bias<<<dim3((B*Na)/128, 4), dim3(256), 0, stream>>>(out_aud, Wv_a, bv_a, bufB);

  // 7) image attends over audio -> fused_i
  flash_attn<<<dim3(Nv/64, H, B), dim3(128), 0, stream>>>(bufA, bufC, bufB,
      bias_i, bufE, Nv, Na, scale);

  // 8) out-proj + gate -> image_out
  gemm_bias<<<dim3((B*Nv)/128, 4), dim3(256), 0, stream>>>(bufE, Wo_i, bo_i, bufA);
  gate_blend<<<dim3((unsigned)((IMGSZ + 255) / 256)), dim3(256), 0, stream>>>(
      bufA, gate_i, image, out_img, (int)IMGSZ, Nv * C, C);

  (void)in_sizes; (void)n_in; (void)out_size; (void)ws_size;
}
